// TransformerARDecoder_18227841204401
// MI455X (gfx1250) — matrix operand order repaired and checked by reading, not compiled, not run on hardware
//
#include <hip/hip_runtime.h>
#include <math.h>

typedef unsigned short u16t;
typedef __attribute__((ext_vector_type(16))) __bf16 v16bf;
typedef __attribute__((ext_vector_type(8)))  float  v8f;

union Frag { v16bf v; uint4 q[2]; };

#define V8Z ((v8f){0.f,0.f,0.f,0.f,0.f,0.f,0.f,0.f})

__device__ __forceinline__ u16t f2bf(float f) {
  union { float f; unsigned u; } x; x.f = f;
  unsigned r = x.u + 0x7FFFu + ((x.u >> 16) & 1u);
  return (u16t)(r >> 16);
}

__device__ __forceinline__ v8f wmma_bf16(const Frag& a, const Frag& b, v8f c) {
  return __builtin_amdgcn_wmma_f32_16x16x32_bf16(false, a.v, false, b.v, (short)0, c, false, false);
}

// A-matrix fragment (16x32 bf16): lane = M row; half=0 -> K {0..7,16..23}, half=1 -> K {8..15,24..31}
__device__ __forceinline__ void loadA(Frag& f, const u16t* row, int k0, int half) {
  f.q[0] = *(const uint4*)(row + k0 + half * 8);
  f.q[1] = *(const uint4*)(row + k0 + 16 + half * 8);
}
// B-matrix fragment (32x16 bf16): lane = N column; half=0 -> K {0..7,16..23}, half=1 -> K {8..15,24..31}
__device__ __forceinline__ void loadB(Frag& f, const u16t* row, int k0, int half) {
  const u16t* p = row + k0 + half * 8;
  f.q[0] = *(const uint4*)(p);
  f.q[1] = *(const uint4*)(p + 16);
}

// ---------------- elementwise conversions ----------------
__global__ __launch_bounds__(256) void conv_kernel(const float* __restrict__ in,
                                                   u16t* __restrict__ out, int n) {
  int i = blockIdx.x * 256 + threadIdx.x;
  if (i < n) out[i] = f2bf(in[i]);
}

// fp32 [K,N] row-major -> bf16 [N,K] row-major (transpose+convert)
__global__ __launch_bounds__(256) void tconv_kernel(const float* __restrict__ in,
                                                    u16t* __restrict__ out,
                                                    int K, int N, int total) {
  int idx = blockIdx.x * 256 + threadIdx.x;
  if (idx >= total) return;
  int n = idx / K, k = idx - n * K;
  out[idx] = f2bf(in[(size_t)k * N + n]);
}

// V^T builder: out[((b*H+h)*HD + c)*Tk + t] = in[(b*Tk+t)*stride + colOff + h*HD + c]
__global__ __launch_bounds__(256) void vtrans_kernel(const u16t* __restrict__ in, int inStride,
                                                     int colOff, u16t* __restrict__ out,
                                                     int H, int HD, int Tk, int total) {
  int idx = blockIdx.x * 256 + threadIdx.x;
  if (idx >= total) return;
  int t = idx % Tk;
  int r = idx / Tk;
  int c = r % HD;
  int bh = r / HD;
  int h = bh % H;
  int b = bh / H;
  out[idx] = in[(size_t)(b * Tk + t) * inStride + colOff + h * HD + c];
}

// ---------------- GEMM: C[M,N] = A[M,K](bf16) x Bt[N,K](bf16) + bias (+relu) (+residual) ----------------
// 4 waves per block; each wave computes a 32(M) x 64(N) tile => block tile 64x128.
// Per 32-K step: 6 x (2x b128) fragment loads feed 8 v_wmma_f32_16x16x32_bf16.
__global__ __launch_bounds__(128) void gemm_bf16_wmma(
    const u16t* __restrict__ A, const u16t* __restrict__ Bt,
    const float* __restrict__ bias, const float* resid,
    float* outF, u16t* outB, int M, int N, int K, int doRelu) {
  const int lane = threadIdx.x & 31;
  const int wave = threadIdx.x >> 5;
  const int m0 = blockIdx.y * 64 + (wave >> 1) * 32;
  const int n0 = blockIdx.x * 128 + (wave & 1) * 64;
  if (m0 >= M || n0 >= N) return;
  const int t16 = lane & 15;
  const int half = lane >> 4;

  v8f acc[2][4];
#pragma unroll
  for (int i = 0; i < 2; ++i)
#pragma unroll
    for (int j = 0; j < 4; ++j) acc[i][j] = V8Z;

  const u16t* ap0 = A + (size_t)(m0 + t16) * K;
  const u16t* ap1 = A + (size_t)(m0 + 16 + t16) * K;
  const u16t* bp0 = Bt + (size_t)(n0 + t16) * K;

  for (int k = 0; k < K; k += 32) {
    Frag a[2], b[4];
    loadA(a[0], ap0, k, half);
    loadA(a[1], ap1, k, half);
#pragma unroll
    for (int j = 0; j < 4; ++j) loadB(b[j], bp0 + (size_t)(j * 16) * K, k, half);
#pragma unroll
    for (int i = 0; i < 2; ++i)
#pragma unroll
      for (int j = 0; j < 4; ++j) acc[i][j] = wmma_bf16(a[i], b[j], acc[i][j]);
  }

#pragma unroll
  for (int i = 0; i < 2; ++i)
#pragma unroll
    for (int j = 0; j < 4; ++j) {
      int col = n0 + j * 16 + t16;
      float bv = bias ? bias[col] : 0.f;
#pragma unroll
      for (int r = 0; r < 8; ++r) {
        int row = m0 + i * 16 + r + half * 8;
        size_t o = (size_t)row * N + col;
        float v = acc[i][j][r] + bv;
        if (doRelu) v = fmaxf(v, 0.f);
        if (resid) v += resid[o];
        if (outF) outF[o] = v;
        if (outB) outB[o] = f2bf(v);
      }
    }
}

// ---------------- LayerNorm (per row) -> bf16 ----------------
__global__ __launch_bounds__(256) void ln_kernel(const float* __restrict__ x,
                                                 const float* __restrict__ s,
                                                 const float* __restrict__ b,
                                                 u16t* __restrict__ out, int D) {
  int row = blockIdx.x;
  const float* xr = x + (size_t)row * D;
  float sum = 0.f, sq = 0.f;
  for (int c = threadIdx.x; c < D; c += 256) { float v = xr[c]; sum += v; sq += v * v; }
  __shared__ float s1[256], s2[256];
  s1[threadIdx.x] = sum; s2[threadIdx.x] = sq;
  __syncthreads();
  for (int o = 128; o > 0; o >>= 1) {
    if ((int)threadIdx.x < o) { s1[threadIdx.x] += s1[threadIdx.x + o]; s2[threadIdx.x] += s2[threadIdx.x + o]; }
    __syncthreads();
  }
  float mu = s1[0] / (float)D;
  float var = s2[0] / (float)D - mu * mu;
  float inv = rsqrtf(var + 1e-5f);
  for (int c = threadIdx.x; c < D; c += 256)
    out[(size_t)row * D + c] = f2bf((xr[c] - mu) * inv * s[c] + b[c]);
}

// ---------------- build x = tgt + positional encoding ----------------
__global__ __launch_bounds__(256) void build_x_kernel(
    const float* __restrict__ tfh, const float* __restrict__ statE,
    const float* __restrict__ startTok, float* __restrict__ x,
    int T, int D, int CD, int SD, float peScale, int total) {
  int idx = blockIdx.x * 256 + threadIdx.x;
  if (idx >= total) return;
  int c = idx % D;
  int bt = idx / D;
  int t = bt % T;
  int b = bt / T;
  float v;
  if (t == 0)        v = startTok[c];
  else if (c < CD)   v = tfh[(size_t)(b * T + (t - 1)) * CD + c];
  else               v = statE[(size_t)b * SD + (c - CD)];
  float ang = (float)t * __expf(-(float)(c & ~1) * peScale);
  v += (c & 1) ? __cosf(ang) : __sinf(ang);
  x[idx] = v;
}

// ---------------- fused attention (flash-style, one wave per 16-query tile, HD==64) ----------------
__global__ __launch_bounds__(128) void attn_kernel(
    const u16t* __restrict__ Q, int qs, int qoff,
    const u16t* __restrict__ Kmat, int ks, int koff,
    const u16t* __restrict__ vT,
    u16t* __restrict__ O, int os, int ooff,
    int H, int Tq, int Tk, int HD,
    int causal, const unsigned char* __restrict__ mask, float scale) {
  const int lane = threadIdx.x & 31;
  const int wave = threadIdx.x >> 5;
  const int qb = (blockIdx.x * 4 + wave) * 16;
  const int h = blockIdx.y;
  const int b = blockIdx.z;
  if (qb >= Tq) return;
  const int t16 = lane & 15;
  const int half = lane >> 4;

  __shared__ __align__(16) u16t Pbuf[4][16 * 32];
  u16t* P = Pbuf[wave];

  const u16t* qrow = Q + (size_t)(b * Tq + qb + t16) * qs + qoff + h * HD;
  Frag qa0, qa1;
  loadA(qa0, qrow, 0, half);
  loadA(qa1, qrow, 32, half);

  const u16t* kbase = Kmat + (size_t)(b * Tk) * ks + koff + h * HD;
  const u16t* vbase = vT + (size_t)((b * H + h) * HD) * Tk;

  v8f o0 = V8Z, o1 = V8Z, o2 = V8Z, o3 = V8Z;
  float rowmax[8], rowsum[8];
#pragma unroll
  for (int r = 0; r < 8; ++r) { rowmax[r] = -1e30f; rowsum[r] = 0.f; }

  const int kend = causal ? (qb + 16) : Tk;
  for (int kk = 0; kk < kend; kk += 32) {
    v8f sv[2];
#pragma unroll
    for (int sub = 0; sub < 2; ++sub) {
      int kt = kk + sub * 16;
      v8f s = V8Z;
      if (kt < kend) {
        const u16t* krow = kbase + (size_t)(kt + t16) * ks;
        Frag kf;
        loadB(kf, krow, 0, half);
        s = wmma_bf16(qa0, kf, s);
        loadB(kf, krow, 32, half);
        s = wmma_bf16(qa1, kf, s);
        float addb = 0.f;
        if (mask) addb = mask[(size_t)b * Tk + kt + t16] ? 0.f : -1e9f;
#pragma unroll
        for (int r = 0; r < 8; ++r) {
          float v = s[r] * scale + addb;
          if (causal && (kt + t16) > (qb + r + half * 8)) v = -1e9f;
          s[r] = v;
        }
      } else {
#pragma unroll
        for (int r = 0; r < 8; ++r) s[r] = -1e9f;
      }
      sv[sub] = s;
    }
    // online softmax update; row (r + 8*half) spans the 16 lanes of this half
#pragma unroll
    for (int r = 0; r < 8; ++r) {
      float mx = fmaxf(sv[0][r], sv[1][r]);
#pragma unroll
      for (int m = 1; m <= 8; m <<= 1) mx = fmaxf(mx, __shfl_xor(mx, m, 32));
      float nm = fmaxf(rowmax[r], mx);
      float corr = __expf(rowmax[r] - nm);
      rowmax[r] = nm;
      float p0 = __expf(sv[0][r] - nm);
      float p1 = __expf(sv[1][r] - nm);
      float ps = p0 + p1;
#pragma unroll
      for (int m = 1; m <= 8; m <<= 1) ps += __shfl_xor(ps, m, 32);
      rowsum[r] = rowsum[r] * corr + ps;
      o0[r] *= corr; o1[r] *= corr; o2[r] *= corr; o3[r] *= corr;
      P[(r + half * 8) * 32 + t16] = f2bf(p0);
      P[(r + half * 8) * 32 + 16 + t16] = f2bf(p1);
    }
    // P (C-layout) -> A-fragment via per-wave LDS tile (DS ops are in-order within a wave)
    Frag pf;
    loadA(pf, P + t16 * 32, 0, half);
    // O += P[16x32] * V[32x64]; vT rows are contiguous over keys
    Frag vf;
    const u16t* vrow = vbase + (size_t)t16 * Tk;
    loadB(vf, vrow, kk, half);                   o0 = wmma_bf16(pf, vf, o0);
    loadB(vf, vrow + (size_t)16 * Tk, kk, half); o1 = wmma_bf16(pf, vf, o1);
    loadB(vf, vrow + (size_t)32 * Tk, kk, half); o2 = wmma_bf16(pf, vf, o2);
    loadB(vf, vrow + (size_t)48 * Tk, kk, half); o3 = wmma_bf16(pf, vf, o3);
  }

  u16t* obase = O + (size_t)(b * Tq + qb) * os + ooff + h * HD;
#pragma unroll
  for (int r = 0; r < 8; ++r) {
    float inv = 1.f / rowsum[r];
    u16t* orow = obase + (size_t)(r + half * 8) * os;
    orow[t16]      = f2bf(o0[r] * inv);
    orow[16 + t16] = f2bf(o1[r] * inv);
    orow[32 + t16] = f2bf(o2[r] * inv);
    orow[48 + t16] = f2bf(o3[r] * inv);
  }
}

// ---------------- host orchestration ----------------
extern "C" void kernel_launch(void* const* d_in, const int* in_sizes, int n_in,
                              void* d_out, int out_size, void* d_ws, size_t ws_size,
                              hipStream_t stream) {
  (void)n_in; (void)out_size;
  const float* context  = (const float*)d_in[0];
  const float* staticE  = (const float*)d_in[1];
  const float* tfIn     = (const float*)d_in[2];
  const unsigned char* amask = (const unsigned char*)d_in[3];
  const float* memW  = (const float*)d_in[5];
  const float* memBi = (const float*)d_in[6];
  const float* tfW   = (const float*)d_in[7];
  const float* tfBi  = (const float*)d_in[8];
  const float* startTok = (const float*)d_in[9];
  const float* ln1s = (const float*)d_in[10]; const float* ln1b = (const float*)d_in[11];
  const float* qkvW = (const float*)d_in[12]; const float* qkvBi = (const float*)d_in[13];
  const float* outsW = (const float*)d_in[14]; const float* outsBi = (const float*)d_in[15];
  const float* ln2s = (const float*)d_in[16]; const float* ln2b = (const float*)d_in[17];
  const float* qcW = (const float*)d_in[18]; const float* qcBi = (const float*)d_in[19];
  const float* kvW = (const float*)d_in[20]; const float* kvBi = (const float*)d_in[21];
  const float* outcW = (const float*)d_in[22]; const float* outcBi = (const float*)d_in[23];
  const float* ln3s = (const float*)d_in[24]; const float* ln3b = (const float*)d_in[25];
  const float* f1W = (const float*)d_in[26]; const float* f1Bi = (const float*)d_in[27];
  const float* f2W = (const float*)d_in[28]; const float* f2Bi = (const float*)d_in[29];
  const float* headW = (const float*)d_in[30]; const float* headBi = (const float*)d_in[31];

  const int D   = in_sizes[6];
  const int CD  = in_sizes[5] / D;
  const int SD  = D - CD;
  const int OD  = in_sizes[31];
  const int B   = in_sizes[1] / SD;
  const int L   = in_sizes[0] / (B * CD);
  const int T   = in_sizes[2] / (B * OD);
  const int NL  = in_sizes[10] / D;
  const int DFF = in_sizes[27] / NL;
  const int H = 16, HD = D / H;
  const int BT = B * T, BL = B * L;
  const int TKMAX = (L > T) ? L : T;

  char* p = (char*)d_ws;
  auto a16 = [&](size_t e) { u16t* r = (u16t*)p; p += ((e * 2 + 255) / 256) * 256; return r; };
  auto aF  = [&](size_t e) { float* r = (float*)p; p += ((e * 4 + 255) / 256) * 256; return r; };

  u16t* ctxB  = a16((size_t)BL * CD);
  u16t* memB  = a16((size_t)BL * D);
  u16t* tfInB = a16((size_t)BT * OD);
  u16t* hB    = a16((size_t)BT * D);
  u16t* qkvB  = a16((size_t)BT * 3 * D);
  u16t* qB    = a16((size_t)BT * D);
  u16t* kvB   = a16((size_t)BL * 2 * D);
  u16t* vTB   = a16((size_t)B * H * HD * TKMAX);
  u16t* attnB = a16((size_t)BT * D);
  u16t* ffnB  = a16((size_t)BT * DFF);
  u16t* xB    = a16((size_t)BT * D);
  u16t* memWt = a16((size_t)D * CD);
  u16t* tfWt  = a16((size_t)CD * OD);
  u16t* qkvWt = a16((size_t)NL * 3 * D * D);
  u16t* outsWt = a16((size_t)NL * D * D);
  u16t* qcWt  = a16((size_t)NL * D * D);
  u16t* kvWt  = a16((size_t)NL * 2 * D * D);
  u16t* outcWt = a16((size_t)NL * D * D);
  u16t* f1Wt  = a16((size_t)NL * DFF * D);
  u16t* f2Wt  = a16((size_t)NL * D * DFF);
  u16t* headWt = a16((size_t)OD * D);
  float* xF  = aF((size_t)BT * D);
  float* tfh = aF((size_t)BT * CD);
  if ((size_t)(p - (char*)d_ws) > ws_size) return;  // not enough scratch

  auto cgrid = [](size_t n) { return dim3((unsigned)((n + 255) / 256)); };
  auto conv = [&](const float* in, u16t* out, size_t n) {
    conv_kernel<<<cgrid(n), 256, 0, stream>>>(in, out, (int)n);
  };
  auto tconv = [&](const float* in, u16t* out, int K, int N) {
    size_t n = (size_t)K * N;
    tconv_kernel<<<cgrid(n), 256, 0, stream>>>(in, out, K, N, (int)n);
  };
  auto gemm = [&](const u16t* A, const u16t* Bt, const float* bias, const float* res,
                  float* oF, u16t* oB, int M, int N, int K, int relu) {
    dim3 g((N + 127) / 128, (M + 63) / 64);
    gemm_bf16_wmma<<<g, 128, 0, stream>>>(A, Bt, bias, res, oF, oB, M, N, K, relu);
  };

  // ---- weight transpose-converts (done on every call; ~70M elems, bandwidth-trivial) ----
  conv(context, ctxB, (size_t)BL * CD);
  conv(tfIn, tfInB, (size_t)BT * OD);
  tconv(memW, memWt, CD, D);
  tconv(tfW, tfWt, OD, CD);
  tconv(headW, headWt, D, OD);
  for (int l = 0; l < NL; ++l) {
    tconv(qkvW + (size_t)l * D * 3 * D, qkvWt + (size_t)l * 3 * D * D, D, 3 * D);
    tconv(outsW + (size_t)l * D * D, outsWt + (size_t)l * D * D, D, D);
    tconv(qcW + (size_t)l * D * D, qcWt + (size_t)l * D * D, D, D);
    tconv(kvW + (size_t)l * D * 2 * D, kvWt + (size_t)l * 2 * D * D, D, 2 * D);
    tconv(outcW + (size_t)l * D * D, outcWt + (size_t)l * D * D, D, D);
    tconv(f1W + (size_t)l * D * DFF, f1Wt + (size_t)l * DFF * D, D, DFF);
    tconv(f2W + (size_t)l * DFF * D, f2Wt + (size_t)l * D * DFF, DFF, D);
  }

  // ---- memory = context @ mem_W + mem_b (bf16 for cross-attn KV projections) ----
  gemm(ctxB, memWt, memBi, nullptr, nullptr, memB, BL, D, CD, 0);
  // ---- tfh = relu(tf_inputs @ tf_W + tf_b) ----
  gemm(tfInB, tfWt, tfBi, nullptr, tfh, nullptr, BT, CD, OD, 1);
  // ---- x = [start | concat(tfh, static)] + PE ----
  {
    size_t n = (size_t)BT * D;
    build_x_kernel<<<cgrid(n), 256, 0, stream>>>(tfh, staticE, startTok, xF, T, D, CD, SD,
                                                 logf(10000.f) / (float)D, (int)n);
  }

  const float scale = 1.f / sqrtf((float)HD);
  for (int l = 0; l < NL; ++l) {
    // self-attention block
    ln_kernel<<<BT, 256, 0, stream>>>(xF, ln1s + (size_t)l * D, ln1b + (size_t)l * D, hB, D);
    gemm(hB, qkvWt + (size_t)l * 3 * D * D, qkvBi + (size_t)l * 3 * D, nullptr, nullptr, qkvB,
         BT, 3 * D, D, 0);
    {
      size_t n = (size_t)B * H * HD * T;
      vtrans_kernel<<<cgrid(n), 256, 0, stream>>>(qkvB, 3 * D, 2 * D, vTB, H, HD, T, (int)n);
    }
    attn_kernel<<<dim3((T + 63) / 64, H, B), 128, 0, stream>>>(
        qkvB, 3 * D, 0, qkvB, 3 * D, D, vTB, attnB, D, 0, H, T, T, HD, 1, nullptr, scale);
    gemm(attnB, outsWt + (size_t)l * D * D, outsBi + (size_t)l * D, xF, xF, nullptr, BT, D, D, 0);

    // cross-attention block
    ln_kernel<<<BT, 256, 0, stream>>>(xF, ln2s + (size_t)l * D, ln2b + (size_t)l * D, hB, D);
    gemm(hB, qcWt + (size_t)l * D * D, qcBi + (size_t)l * D, nullptr, nullptr, qB, BT, D, D, 0);
    gemm(memB, kvWt + (size_t)l * 2 * D * D, kvBi + (size_t)l * 2 * D, nullptr, nullptr, kvB,
         BL, 2 * D, D, 0);
    {
      size_t n = (size_t)B * H * HD * L;
      vtrans_kernel<<<cgrid(n), 256, 0, stream>>>(kvB, 2 * D, D, vTB, H, HD, L, (int)n);
    }
    attn_kernel<<<dim3((T + 63) / 64, H, B), 128, 0, stream>>>(
        qB, D, 0, kvB, 2 * D, 0, vTB, attnB, D, 0, H, T, L, HD, 0, amask, scale);
    gemm(attnB, outcWt + (size_t)l * D * D, outcBi + (size_t)l * D, xF, xF, nullptr, BT, D, D, 0);

    // FFN block
    ln_kernel<<<BT, 256, 0, stream>>>(xF, ln3s + (size_t)l * D, ln3b + (size_t)l * D, hB, D);
    gemm(hB, f1Wt + (size_t)l * DFF * D, f1Bi + (size_t)l * DFF, nullptr, nullptr, ffnB,
         BT, DFF, D, 1);
    gemm(ffnB, f2Wt + (size_t)l * D * DFF, f2Bi + (size_t)l * D, xF, xF, nullptr, BT, D, DFF, 0);
  }

  // ---- head: out = x @ head_W + head_b ----
  conv(xF, xB, (size_t)BT * D);
  gemm(xB, headWt, headBi, nullptr, (float*)d_out, nullptr, BT, OD, D, 0);
}
